// EchoStateNetwork_11364483465314
// MI455X (gfx1250) — matrix operand order repaired and checked by reading, not compiled, not run on hardware
//
#include <hip/hip_runtime.h>
#include <hip/hip_bf16.h>

// ESN recurrence on gfx1250: persistent 16-WG kernel, W_res resident in LDS
// as pre-swizzled WMMA-B fragments (bf16 hi+lo), compensated-bf16 WMMA
// (3x v_wmma_f32_16x16x32_bf16 per 32-wide K chunk), grid sync via global
// atomic barrier, cross-WGP state exchange through a ping-pong hi/lo bf16
// buffer in d_ws.

typedef __attribute__((ext_vector_type(16))) __bf16 bf16x16;
typedef __attribute__((ext_vector_type(8)))  float  f32x8;

#define BATCH   32
#define TSTEPS  2048
#define IN_DIM  64
#define RES     1024
#define NWG     16
#define SLICE   64              // output columns per workgroup
#define KCH_IN  2               // K chunks (of 32) covering the input part
#define KCH_ST  32              // K chunks covering the state part
#define KCH     (KCH_IN + KCH_ST)
#define NTILES  4               // 16-wide N tiles per WG slice

union FragU {
    bf16x16 v;
    uint4   q[2];
};

static __device__ __forceinline__ unsigned short f2bf(float f) {
    __bf16 h = (__bf16)f;
    return __builtin_bit_cast(unsigned short, h);
}
static __device__ __forceinline__ float bf2f(unsigned short u) {
    __bf16 h = __builtin_bit_cast(__bf16, u);
    return (float)h;
}

static __device__ __forceinline__ void cvt4(float4 f, FragU& ah, FragU& al, int base) {
    {
        __bf16 h = (__bf16)f.x; ah.v[base + 0] = h; al.v[base + 0] = (__bf16)(f.x - (float)h);
    }
    {
        __bf16 h = (__bf16)f.y; ah.v[base + 1] = h; al.v[base + 1] = (__bf16)(f.y - (float)h);
    }
    {
        __bf16 h = (__bf16)f.z; ah.v[base + 2] = h; al.v[base + 2] = (__bf16)(f.z - (float)h);
    }
    {
        __bf16 h = (__bf16)f.w; ah.v[base + 3] = h; al.v[base + 3] = (__bf16)(f.w - (float)h);
    }
}

static __device__ __forceinline__ f32x8 wmma_bf16(const bf16x16& a, const bf16x16& b, f32x8 c) {
    return __builtin_amdgcn_wmma_f32_16x16x32_bf16(
        /*neg_a=*/false, a, /*neg_b=*/false, b,
        /*c_mod=*/(short)0, c, /*reuse_a=*/false, /*reuse_b=*/false);
}

// LDS: B fragments, pre-swizzled to the wave32 WMMA B layout.
// frag element offset: ((chunk*NTILES + ntile)*32 + lane)*16 + e
// lane(0-15)=column n0+lane, K = kbase + e (e < 8), kbase + 8 + e (e >= 8)
// lane(16-31)=column n0+lane-16, K = kbase + 8 + e (e < 8), kbase + 16 + e (e >= 8)
__global__ __launch_bounds__(256, 1)
void esn_persistent_kernel(const float* __restrict__ in,     // [32,2048,64]
                           const float* __restrict__ Win,    // [64,1024]
                           const float* __restrict__ Wres,   // [1024,1024]
                           float* __restrict__ out,          // [32,2048,1024]
                           unsigned* __restrict__ ctr,
                           unsigned short* __restrict__ exch) // 2 bufs x (hi,lo) x [32,1024]
{
    __shared__ unsigned short sBhi[KCH * NTILES * 32 * 16]; // 69632 elems = 136 KB
    __shared__ unsigned short sBlo[KCH * NTILES * 32 * 16]; // 136 KB

    const int tid = threadIdx.x;
    const int wg  = blockIdx.x;

    // ---- one-time fill: W_in|W_res column slab -> swizzled hi/lo bf16 B frags
    for (int idx = tid; idx < 1088 * SLICE; idx += 256) {
        const int K  = idx >> 6;          // 0..1087 (0..63 = W_in rows, 64.. = W_res rows)
        const int nl = idx & 63;          // local column
        const float w = (K < IN_DIM) ? Win[K * RES + wg * SLICE + nl]
                                     : Wres[(K - IN_DIM) * RES + wg * SLICE + nl];
        const unsigned short h = f2bf(w);
        const unsigned short l = f2bf(w - bf2f(h));
        const int c    = K >> 5;
        const int rem  = K & 31;
        const int sel  = (rem >> 3) & 1;
        const int e    = (rem & 7) | ((rem >> 4) << 3);
        const int lane = sel * 16 + (nl & 15);
        const int nt   = nl >> 4;
        const int off  = ((c * NTILES + nt) * 32 + lane) * 16 + e;
        sBhi[off] = h;
        sBlo[off] = l;
    }
    __syncthreads();

    const uint4* sBhi4 = (const uint4*)sBhi;
    const uint4* sBlo4 = (const uint4*)sBlo;

    const int wave   = tid >> 5;
    const int lane   = tid & 31;
    const int mtile  = wave & 1;          // rows 0-15 / 16-31
    const int ntile  = wave >> 1;         // 0..3 within the 64-col slice
    const int sel    = lane >> 4;
    const int nloc   = lane & 15;
    const int mrow   = mtile * 16 + nloc; // A-source batch row for this lane
    const int Nabs   = wg * SLICE + ntile * 16 + nloc;

    f32x8 prev;
    #pragma unroll
    for (int g = 0; g < 8; ++g) prev[g] = 0.0f;

    for (int t = 0; t < TSTEPS; ++t) {
        f32x8 acc;
        #pragma unroll
        for (int g = 0; g < 8; ++g) acc[g] = 0.0f;

        // ---- input-projection part (K = 0..63), fused: A from inputs[:,t,:]
        const float* inrow = in + ((size_t)mrow * TSTEPS + t) * IN_DIM;
        #pragma unroll
        for (int c = 0; c < KCH_IN; ++c) {
            const int k0 = c * 32 + sel * 8;
            FragU ah, al;
            cvt4(*(const float4*)(inrow + k0),      ah, al, 0);
            cvt4(*(const float4*)(inrow + k0 + 4),  ah, al, 4);
            cvt4(*(const float4*)(inrow + k0 + 16), ah, al, 8);
            cvt4(*(const float4*)(inrow + k0 + 20), ah, al, 12);
            FragU bh, bl;
            const int fb = ((c * NTILES + ntile) * 32 + lane) * 2;
            bh.q[0] = sBhi4[fb]; bh.q[1] = sBhi4[fb + 1];
            bl.q[0] = sBlo4[fb]; bl.q[1] = sBlo4[fb + 1];
            acc = wmma_bf16(ah.v, bh.v, acc);
            acc = wmma_bf16(ah.v, bl.v, acc);
            acc = wmma_bf16(al.v, bh.v, acc);
        }

        // ---- recurrent part (K = 64..1087): A from hi/lo bf16 exchange buffer
        if (t > 0) {
            const int rbuf = (t - 1) & 1;
            const unsigned short* exh = exch + (rbuf * 2 + 0) * (BATCH * RES) + mrow * RES;
            const unsigned short* exl = exch + (rbuf * 2 + 1) * (BATCH * RES) + mrow * RES;
            #pragma unroll 4
            for (int c = 0; c < KCH_ST; ++c) {
                const int k0 = c * 32 + sel * 8;
                FragU ah, al, bh, bl;
                ah.q[0] = *(const uint4*)(exh + k0);
                ah.q[1] = *(const uint4*)(exh + k0 + 16);
                al.q[0] = *(const uint4*)(exl + k0);
                al.q[1] = *(const uint4*)(exl + k0 + 16);
                const int fb = (((c + KCH_IN) * NTILES + ntile) * 32 + lane) * 2;
                bh.q[0] = sBhi4[fb]; bh.q[1] = sBhi4[fb + 1];
                bl.q[0] = sBlo4[fb]; bl.q[1] = sBlo4[fb + 1];
                acc = wmma_bf16(ah.v, bh.v, acc);
                acc = wmma_bf16(ah.v, bl.v, acc);
                acc = wmma_bf16(al.v, bh.v, acc);
            }
        }

        // ---- epilogue: tanh + leaky update; write fp32 output + hi/lo exchange
        const int wbuf = t & 1;
        unsigned short* wxh = exch + (wbuf * 2 + 0) * (BATCH * RES);
        unsigned short* wxl = exch + (wbuf * 2 + 1) * (BATCH * RES);
        #pragma unroll
        for (int g = 0; g < 8; ++g) {
            const int M = mtile * 16 + sel * 8 + g;
            const float z  = acc[g];
            const float nv = tanhf(z);
            const float o  = 0.7f * prev[g] + 0.3f * nv;
            prev[g] = o;
            out[((size_t)M * TSTEPS + t) * RES + Nabs] = o;
            const unsigned short h = f2bf(o);
            wxh[M * RES + Nabs] = h;
            wxl[M * RES + Nabs] = f2bf(o - bf2f(h));
        }

        // ---- grid-wide release/acquire barrier (persistent-kernel sync)
        if (t != TSTEPS - 1) {
            __threadfence();          // make this wave's stores device-visible
            __syncthreads();
            if (tid == 0) {
                __hip_atomic_fetch_add(ctr, 1u, __ATOMIC_RELEASE, __HIP_MEMORY_SCOPE_AGENT);
                const unsigned tgt = (unsigned)(NWG * (t + 1));
                while (__hip_atomic_load(ctr, __ATOMIC_ACQUIRE, __HIP_MEMORY_SCOPE_AGENT) < tgt) {
                    __builtin_amdgcn_s_sleep(1);
                }
            }
            __syncthreads();          // acquire-inv done by thread 0 releases the WG
        }
    }
}

extern "C" void kernel_launch(void* const* d_in, const int* in_sizes, int n_in,
                              void* d_out, int out_size, void* d_ws, size_t ws_size,
                              hipStream_t stream) {
    (void)in_sizes; (void)n_in; (void)out_size; (void)ws_size;
    const float* in   = (const float*)d_in[0];
    const float* Win  = (const float*)d_in[1];
    const float* Wres = (const float*)d_in[2];
    float* out = (float*)d_out;

    unsigned*       ctr  = (unsigned*)d_ws;
    unsigned short* exch = (unsigned short*)((char*)d_ws + 4096); // 256 KB used

    hipMemsetAsync(d_ws, 0, 4096, stream);  // zero barrier counter (graph-safe)
    hipLaunchKernelGGL(esn_persistent_kernel, dim3(NWG), dim3(256), 0, stream,
                       in, Win, Wres, out, ctr, exch);
}
